// SymODE_54614804135957
// MI455X (gfx1250) — compile-verified, matrix operand order checked by reading, not run on hardware
//
#include <hip/hip_runtime.h>
#include <hip/hip_bf16.h>

// ---------------------------------------------------------------------------
// SymODE for MI455X (gfx1250): batch-tiled WMMA implementation.
// Each workgroup = 16 trajectories, 8 wave32.  Every MLP layer (forward,
// backward-gated, JVP-gated) is a (16 x K) * (K x N) GEMM executed with
// v_wmma_f32_16x16x32_f16.  Weights are pre-packed to f16 fragment order and
// streamed from L2; activations/gates live in LDS (ds_load_b128 fragments).
// Epilogues use native v_exp/v_log/v_rcp transcendentals.
// ---------------------------------------------------------------------------

typedef __attribute__((ext_vector_type(16))) _Float16 v16h;
typedef __attribute__((ext_vector_type(8)))  _Float16 v8h;
typedef __attribute__((ext_vector_type(8)))  float    v8f;

#define STRIDE_M 416   // padded K/N stride (halves) for the 400-wide M-net
#define STRIDE_V 320   // padded stride for the 300-wide V/G nets

#define LOG2E 1.4426950408889634f
#define LN2   0.6931471805599453f

enum GemmMode { MODE_SOFTPLUS = 0, MODE_GATED = 1, MODE_LIN_SMALL = 2 };

struct Params {
  const float* x;
  const float* times;
  const _Float16 *mw1, *mw2, *mw3, *mw4, *mw4t, *mw3t, *mw2t, *mw1t;
  const _Float16 *vw1, *vw2, *vw3t, *vw2t, *vw1t;
  const _Float16 *gw1, *gw2, *gw3;
  const float *bm1, *bm2, *bm3, *bm4, *bv1, *bv2, *bg1, *bg2, *bg3;
  float* partial;
};

// ---------------------------------------------------------------------------
// Block-wide GEMM: OUT(16 x Ntiles*16) = ACT(16 x Ktiles*32) @ Wblob (+bias)
// Wblob is fragment-ordered f16: block (nt*Ktiles + kt) is 1KB, lane l owns
// halves [l*16 .. l*16+15] = its v16h B fragment.  A fragments come from LDS.
// Fully inlined into the kernel so address-space inference turns activation
// traffic into ds_load_b128 / ds_store_b16 and weight traffic into
// global_load_b128 + global_prefetch_b8.
// ---------------------------------------------------------------------------
__device__ __forceinline__ void gemm_tile(
    const _Float16* actin, int instride,
    const _Float16* wblob, const float* bias,
    _Float16* actout, int outstride, _Float16* gate,
    float* outsmall, int Ktiles, int Ntiles, int Nact, int mode) {
  const int tid   = threadIdx.x;
  const int wv    = tid >> 5;
  const int ln    = tid & 31;
  const int nlane = ln & 15;
  const int roff  = (ln >> 4) << 3;   // 0 or 8: row / K sub-offset per half-wave
  const int arow  = nlane;

  for (int nt = wv; nt < Ntiles; nt += 8) {
    v8f c;
    const float bv = bias ? bias[nt * 16 + nlane] : 0.0f;
#pragma unroll
    for (int i = 0; i < 8; ++i) c[i] = bv;

    const _Float16* wb = wblob + (((size_t)nt * (size_t)Ktiles) << 9) + (ln << 4);
    for (int kt = 0; kt < Ktiles; ++kt) {
      const _Float16* ap = actin + arow * instride + (kt << 5) + roff;
      v8h a0 = *(const v8h*)(ap);
      v8h a1 = *(const v8h*)(ap + 16);
      v16h a;
#pragma unroll
      for (int i = 0; i < 8; ++i) { a[i] = a0[i]; a[i + 8] = a1[i]; }
      v16h b = *(const v16h*)(wb + ((size_t)kt << 9));
      __builtin_prefetch(wb + (((size_t)kt + 1) << 9), 0, 1);  // global_prefetch_b8
      c = __builtin_amdgcn_wmma_f32_16x16x32_f16(false, a, false, b,
                                                 (short)0, c, false, false);
    }

    const int gn = nt * 16 + nlane;
    if (mode == MODE_LIN_SMALL) {
      if (gn < 16) {
#pragma unroll
        for (int r = 0; r < 8; ++r) outsmall[(roff + r) * 16 + gn] = c[r];
      }
    } else if (mode == MODE_SOFTPLUS) {
      const bool valid = gn < Nact;
#pragma unroll
      for (int r = 0; r < 8; ++r) {
        const int mr = roff + r;
        float y = c[r];
        // sigmoid(y) = rcp(1 + exp2(-y*log2e))   (v_exp_f32 + v_rcp_f32)
        float sg = __builtin_amdgcn_rcpf(1.0f + __builtin_amdgcn_exp2f(-y * LOG2E));
        // softplus(y) = ln2 * log2(1 + exp2(y*log2e)), clamped for large y
        float e  = __builtin_amdgcn_exp2f(y * LOG2E);
        float sp = LN2 * __builtin_amdgcn_logf(1.0f + e);
        sp = (y > 20.0f) ? y : sp;           // branchless v_cndmask
        actout[mr * outstride + gn] = valid ? (_Float16)sp : (_Float16)0.0f;
        gate[mr * outstride + gn]   = valid ? (_Float16)sg : (_Float16)0.0f;
      }
    } else {  // MODE_GATED: out = (act@W) * stored_sigmoid  (backward / JVP)
      const bool valid = gn < Nact;
#pragma unroll
      for (int r = 0; r < 8; ++r) {
        const int mr = roff + r;
        float g = (float)gate[mr * outstride + gn];
        float y = c[r] * g;
        actout[mr * outstride + gn] = valid ? (_Float16)y : (_Float16)0.0f;
      }
    }
  }
}

__device__ __forceinline__ void seed_clear32(_Float16* buf, int stride, int row) {
#pragma unroll
  for (int j = 0; j < 32; ++j) buf[row * stride + j] = (_Float16)0.0f;
}

// ---------------------------------------------------------------------------
// Main kernel: RK4 with 2 substeps per interval; 16 trajectories per block.
// Single f_eval call site (rolled stage loop) keeps one inlined copy of the
// 20-GEMM RHS evaluator.
// ---------------------------------------------------------------------------
#define SMEM_HALVES (16 * STRIDE_M * 5 + 16 * STRIDE_V * 4)
#define SMEM_F32    (5 * 256 + 4 * 128 + 32 + 32 + 48)
#define SMEM_BYTES  (SMEM_HALVES * 2 + SMEM_F32 * 4)

__global__ void __launch_bounds__(256) symode_main(Params P) {
  extern __shared__ __align__(16) unsigned char smem_raw[];
  _Float16* hp = (_Float16*)smem_raw;
  _Float16* Pb  = hp; hp += 16 * STRIDE_M;
  _Float16* Qb  = hp; hp += 16 * STRIDE_M;
  _Float16* gM1 = hp; hp += 16 * STRIDE_M;
  _Float16* gM2 = hp; hp += 16 * STRIDE_M;
  _Float16* gM3 = hp; hp += 16 * STRIDE_M;
  _Float16* Rb  = hp; hp += 16 * STRIDE_V;
  _Float16* Sb  = hp; hp += 16 * STRIDE_V;
  _Float16* gV1 = hp; hp += 16 * STRIDE_V;
  _Float16* gV2 = hp; hp += 16 * STRIDE_V;
  float* fp = (float*)hp;
  float* mO  = fp; fp += 256;
  float* dcM = fp; fp += 256;
  float* dcV = fp; fp += 256;
  float* gO  = fp; fp += 256;
  float* tO  = fp; fp += 256;
  float* z    = fp; fp += 128;
  float* ycur = fp; fp += 128;
  float* fout = fp; fp += 128;
  float* kacc = fp; fp += 128;
  float* pb  = fp; fp += 32;
  float* dpb = fp; fp += 32;
  float* mi  = fp; fp += 48;

  const int tid = threadIdx.x;
  {  // zero all half buffers once: keeps K-padding regions zero forever
    _Float16* base = (_Float16*)smem_raw;
    for (int i = tid; i < SMEM_HALVES; i += 256) base[i] = (_Float16)0.0f;
  }
  const int traj = blockIdx.x * 16;
  if (tid < 16) {
#pragma unroll
    for (int j = 0; j < 8; ++j)
      z[tid * 8 + j] = P.x[(size_t)(traj + tid) * 80 + j];
  }
  __syncthreads();

  // ---- ODE RHS: fout = f(ycur) for the block's 16 trajectories ----
  auto f_eval = [&]() {
    // M-net forward: c -> h1 -> h2 -> h3 -> o (store sigmoid gates)
    if (tid < 16) {
      seed_clear32(Pb, STRIDE_M, tid);
#pragma unroll
      for (int j = 0; j < 4; ++j)
        Pb[tid * STRIDE_M + j] = (_Float16)ycur[tid * 8 + j];
    }
    __syncthreads();
    gemm_tile(Pb, STRIDE_M, P.mw1, P.bm1, Qb, STRIDE_M, gM1, nullptr, 1, 25, 400, MODE_SOFTPLUS);
    __syncthreads();
    gemm_tile(Qb, STRIDE_M, P.mw2, P.bm2, Pb, STRIDE_M, gM2, nullptr, 13, 25, 400, MODE_SOFTPLUS);
    __syncthreads();
    gemm_tile(Pb, STRIDE_M, P.mw3, P.bm3, Qb, STRIDE_M, gM3, nullptr, 13, 25, 400, MODE_SOFTPLUS);
    __syncthreads();
    gemm_tile(Qb, STRIDE_M, P.mw4, P.bm4, nullptr, 0, nullptr, mO, 13, 1, 16, MODE_LIN_SMALL);
    __syncthreads();

    // scalar: Minv = L L^T + 0.1 I; p = Minv^-1 qdot; cotangent on o
    if (tid < 16) {
      const int r = tid;
      float o0 = mO[r * 16 + 0], o1 = mO[r * 16 + 1], o2 = mO[r * 16 + 2];
      float m00 = o0 * o0 + 0.1f;
      float m01 = o0 * o2;
      float m11 = o1 * o1 + o2 * o2 + 0.1f;
      float inv = __builtin_amdgcn_rcpf(m00 * m11 - m01 * m01);
      float q0 = ycur[r * 8 + 4], q1 = ycur[r * 8 + 5];
      float p0 = (m11 * q0 - m01 * q1) * inv;
      float p1 = (-m01 * q0 + m00 * q1) * inv;
      pb[r * 2 + 0] = p0; pb[r * 2 + 1] = p1;
      mi[r * 3 + 0] = m00; mi[r * 3 + 1] = m01; mi[r * 3 + 2] = m11;
      float do0 = p0 * p0 * o0 + p0 * p1 * o2;   // dH/do for 0.5 p^T Minv(o) p
      float do1 = p1 * p1 * o1;
      float do2 = p0 * p1 * o0 + p1 * p1 * o2;
      seed_clear32(Pb, STRIDE_M, r);
      Pb[r * STRIDE_M + 0] = (_Float16)do0;
      Pb[r * STRIDE_M + 1] = (_Float16)do1;
      Pb[r * STRIDE_M + 2] = (_Float16)do2;
    }
    __syncthreads();

    // M-net backward (gated by stored sigmoids) -> dcM
    gemm_tile(Pb, STRIDE_M, P.mw4t, nullptr, Qb, STRIDE_M, gM3, nullptr, 1, 25, 400, MODE_GATED);
    __syncthreads();
    gemm_tile(Qb, STRIDE_M, P.mw3t, nullptr, Pb, STRIDE_M, gM2, nullptr, 13, 25, 400, MODE_GATED);
    __syncthreads();
    gemm_tile(Pb, STRIDE_M, P.mw2t, nullptr, Qb, STRIDE_M, gM1, nullptr, 13, 25, 400, MODE_GATED);
    __syncthreads();
    gemm_tile(Qb, STRIDE_M, P.mw1t, nullptr, nullptr, 0, nullptr, dcM, 13, 1, 16, MODE_LIN_SMALL);
    __syncthreads();

    // V-net forward (gates only) + backward -> dcV
    if (tid < 16) {
      seed_clear32(Rb, STRIDE_V, tid);
#pragma unroll
      for (int j = 0; j < 4; ++j)
        Rb[tid * STRIDE_V + j] = (_Float16)ycur[tid * 8 + j];
    }
    __syncthreads();
    gemm_tile(Rb, STRIDE_V, P.vw1, P.bv1, Sb, STRIDE_V, gV1, nullptr, 1, 19, 300, MODE_SOFTPLUS);
    __syncthreads();
    gemm_tile(Sb, STRIDE_V, P.vw2, P.bv2, Rb, STRIDE_V, gV2, nullptr, 10, 19, 300, MODE_SOFTPLUS);
    __syncthreads();
    if (tid < 16) {  // seed e0 (cotangent of scalar output)
      seed_clear32(Sb, STRIDE_V, tid);
      Sb[tid * STRIDE_V + 0] = (_Float16)1.0f;
    }
    __syncthreads();
    gemm_tile(Sb, STRIDE_V, P.vw3t, nullptr, Rb, STRIDE_V, gV2, nullptr, 1, 19, 300, MODE_GATED);
    __syncthreads();
    gemm_tile(Rb, STRIDE_V, P.vw2t, nullptr, Sb, STRIDE_V, gV1, nullptr, 10, 19, 300, MODE_GATED);
    __syncthreads();
    gemm_tile(Sb, STRIDE_V, P.vw1t, nullptr, nullptr, 0, nullptr, dcV, 10, 1, 16, MODE_LIN_SMALL);
    __syncthreads();

    // G-net forward (reuses V buffers/gates) -> F matrix
    if (tid < 16) {
      seed_clear32(Rb, STRIDE_V, tid);
#pragma unroll
      for (int j = 0; j < 4; ++j)
        Rb[tid * STRIDE_V + j] = (_Float16)ycur[tid * 8 + j];
    }
    __syncthreads();
    gemm_tile(Rb, STRIDE_V, P.gw1, P.bg1, Sb, STRIDE_V, gV1, nullptr, 1, 19, 300, MODE_SOFTPLUS);
    __syncthreads();
    gemm_tile(Sb, STRIDE_V, P.gw2, P.bg2, Rb, STRIDE_V, gV2, nullptr, 10, 19, 300, MODE_SOFTPLUS);
    __syncthreads();
    gemm_tile(Rb, STRIDE_V, P.gw3, P.bg3, nullptr, 0, nullptr, gO, 10, 1, 16, MODE_LIN_SMALL);
    __syncthreads();

    // scalar: assemble dq, dp, dcqsq; seed JVP tangent
    if (tid < 16) {
      const int r = tid;
      float dh0 = dcM[r * 16 + 0] + dcV[r * 16 + 0];
      float dh1 = dcM[r * 16 + 1] + dcV[r * 16 + 1];
      float dh2 = dcM[r * 16 + 2] + dcV[r * 16 + 2];
      float dh3 = dcM[r * 16 + 3] + dcV[r * 16 + 3];
      float c0 = ycur[r * 8 + 0], c1 = ycur[r * 8 + 1];
      float s0 = ycur[r * 8 + 2], s1 = ycur[r * 8 + 3];
      float u0 = ycur[r * 8 + 6], u1 = ycur[r * 8 + 7];
      float F0 = gO[r * 16 + 0] * u0 + gO[r * 16 + 1] * u1;
      float F1 = gO[r * 16 + 2] * u0 + gO[r * 16 + 3] * u1;
      float dq0 = ycur[r * 8 + 4], dq1 = ycur[r * 8 + 5];  // dH/dp = Minv p = qdot
      float dp0 = s0 * dh0 - c0 * dh2 + F0;
      float dp1 = s1 * dh1 - c1 * dh3 + F1;
      dpb[r * 2 + 0] = dp0; dpb[r * 2 + 1] = dp1;
      float dc0 = -s0 * dq0, dc1 = -s1 * dq1, dc2 = c0 * dq0, dc3 = c1 * dq1;
      fout[r * 8 + 0] = dc0; fout[r * 8 + 1] = dc1;
      fout[r * 8 + 2] = dc2; fout[r * 8 + 3] = dc3;
      fout[r * 8 + 6] = 0.0f; fout[r * 8 + 7] = 0.0f;
      seed_clear32(Pb, STRIDE_M, r);
      Pb[r * STRIDE_M + 0] = (_Float16)dc0;
      Pb[r * STRIDE_M + 1] = (_Float16)dc1;
      Pb[r * STRIDE_M + 2] = (_Float16)dc2;
      Pb[r * STRIDE_M + 3] = (_Float16)dc3;
    }
    __syncthreads();

    // M-net JVP (tangent forward, gated) -> tO
    gemm_tile(Pb, STRIDE_M, P.mw1, nullptr, Qb, STRIDE_M, gM1, nullptr, 1, 25, 400, MODE_GATED);
    __syncthreads();
    gemm_tile(Qb, STRIDE_M, P.mw2, nullptr, Pb, STRIDE_M, gM2, nullptr, 13, 25, 400, MODE_GATED);
    __syncthreads();
    gemm_tile(Pb, STRIDE_M, P.mw3, nullptr, Qb, STRIDE_M, gM3, nullptr, 13, 25, 400, MODE_GATED);
    __syncthreads();
    gemm_tile(Qb, STRIDE_M, P.mw4, nullptr, nullptr, 0, nullptr, tO, 13, 1, 16, MODE_LIN_SMALL);
    __syncthreads();

    // scalar: ddq = Minv dp + dMinv_dt p
    if (tid < 16) {
      const int r = tid;
      float o0 = mO[r * 16 + 0], o1 = mO[r * 16 + 1], o2 = mO[r * 16 + 2];
      float t0 = tO[r * 16 + 0], t1 = tO[r * 16 + 1], t2 = tO[r * 16 + 2];
      float dm00 = 2.0f * o0 * t0;
      float dm01 = t0 * o2 + o0 * t2;
      float dm11 = 2.0f * o1 * t1 + 2.0f * o2 * t2;
      float m00 = mi[r * 3 + 0], m01 = mi[r * 3 + 1], m11 = mi[r * 3 + 2];
      float p0 = pb[r * 2 + 0], p1 = pb[r * 2 + 1];
      float dp0 = dpb[r * 2 + 0], dp1 = dpb[r * 2 + 1];
      fout[r * 8 + 4] = m00 * dp0 + m01 * dp1 + dm00 * p0 + dm01 * p1;
      fout[r * 8 + 5] = m01 * dp0 + m11 * dp1 + dm01 * p0 + dm11 * p1;
    }
    __syncthreads();
  };

  float esum = 0.0f;
#pragma clang loop unroll(disable)
  for (int ti = 0; ti < 9; ++ti) {
#pragma clang loop unroll(disable)
    for (int sub = 0; sub < 2; ++sub) {
      if (tid < 16) {
        for (int j = 0; j < 8; ++j) {
          ycur[tid * 8 + j] = z[tid * 8 + j];
          kacc[tid * 8 + j] = 0.0f;
        }
      }
      __syncthreads();
#pragma clang loop unroll(disable)
      for (int st = 0; st < 4; ++st) {   // single f_eval call site
        f_eval();
        if (tid < 16) {
          const float dt = (P.times[ti + 1] - P.times[ti]) * 0.5f;  // N_SUB=2
          const float wc = (st == 1 || st == 2) ? 2.0f : 1.0f;
          const float ac = (st == 2) ? 1.0f : 0.5f;  // step fraction to next stage
          for (int j = 0; j < 8; ++j) {
            float k = fout[tid * 8 + j];
            kacc[tid * 8 + j] += wc * k;
            if (st < 3)
              ycur[tid * 8 + j] = z[tid * 8 + j] + ac * dt * k;
            else
              z[tid * 8 + j] += (dt / 6.0f) * kacc[tid * 8 + j];
          }
        }
        __syncthreads();
      }
    }
    if (tid < 16) {
      for (int j = 0; j < 8; ++j) {
        float d = P.x[(size_t)(traj + tid) * 80 + (size_t)(ti + 1) * 8 + j] -
                  z[tid * 8 + j];
        esum += d * d;
      }
    }
  }
  if (tid < 16) mO[tid] = esum;
  __syncthreads();
  if (tid == 0) {
    float t = 0.0f;
    for (int i = 0; i < 16; ++i) t += mO[i];
    P.partial[blockIdx.x] = t;
  }
}

// ---------------------------------------------------------------------------
// Weight packing: f32 row-major (optionally transposed) -> f16 fragment blob.
// ---------------------------------------------------------------------------
__global__ void pack_frag(const float* __restrict__ W, int Wrows, int Wcols,
                          int transpose, int Ktiles, int Ntiles,
                          _Float16* __restrict__ dst) {
  int idx = blockIdx.x * 256 + threadIdx.x;
  int total = Ktiles * Ntiles * 512;
  if (idx >= total) return;
  int blk = idx >> 9;
  int within = idx & 511;
  int lane = within >> 4;
  int h = within & 15;
  int nt = blk / Ktiles;
  int kt = blk - nt * Ktiles;
  int n = nt * 16 + (lane & 15);
  int koff = (lane >> 4) << 3;
  int k = kt * 32 + ((h < 8) ? (koff + h) : (16 + koff + (h - 8)));
  float v = 0.0f;
  if (!transpose) {
    if (k < Wrows && n < Wcols) v = W[k * Wcols + n];
  } else {
    if (k < Wcols && n < Wrows) v = W[n * Wcols + k];
  }
  dst[idx] = (_Float16)v;
}

__global__ void pack_bias(const float* __restrict__ b, int nact, int npad,
                          float* __restrict__ dst) {
  int i = blockIdx.x * 256 + threadIdx.x;
  if (i < npad) dst[i] = (i < nact) ? b[i] : 0.0f;
}

__global__ void reduce_partials(const float* __restrict__ partial,
                                float* __restrict__ out, int n, float scale) {
  if (blockIdx.x == 0 && threadIdx.x == 0) {
    float s = 0.0f;
    for (int i = 0; i < n; ++i) s += partial[i];
    out[0] = s * scale;
  }
}

// ---------------------------------------------------------------------------
extern "C" void kernel_launch(void* const* d_in, const int* in_sizes, int n_in,
                              void* d_out, int out_size, void* d_ws,
                              size_t ws_size, hipStream_t stream) {
  (void)in_sizes; (void)n_in; (void)out_size; (void)ws_size;
  const float* x     = (const float*)d_in[0];
  const float* times = (const float*)d_in[1];
  const float* MW[4] = {(const float*)d_in[2], (const float*)d_in[4],
                        (const float*)d_in[6], (const float*)d_in[8]};
  const float* Mb[4] = {(const float*)d_in[3], (const float*)d_in[5],
                        (const float*)d_in[7], (const float*)d_in[9]};
  const float* VW[3] = {(const float*)d_in[10], (const float*)d_in[12],
                        (const float*)d_in[14]};
  const float* Vb[3] = {(const float*)d_in[11], (const float*)d_in[13],
                        (const float*)d_in[15]};
  const float* GW[3] = {(const float*)d_in[16], (const float*)d_in[18],
                        (const float*)d_in[20]};
  const float* Gb[3] = {(const float*)d_in[17], (const float*)d_in[19],
                        (const float*)d_in[21]};

  unsigned char* ws = (unsigned char*)d_ws;
  size_t off = 0;
  auto halloc = [&](size_t halves) -> _Float16* {
    _Float16* p = (_Float16*)(ws + off);
    off += halves * 2;
    off = (off + 255) & ~(size_t)255;
    return p;
  };
  auto falloc = [&](size_t n) -> float* {
    float* p = (float*)(ws + off);
    off += n * 4;
    off = (off + 255) & ~(size_t)255;
    return p;
  };

  _Float16* mw1  = halloc((size_t)1 * 25 * 512);
  _Float16* mw2  = halloc((size_t)13 * 25 * 512);
  _Float16* mw3  = halloc((size_t)13 * 25 * 512);
  _Float16* mw4  = halloc((size_t)13 * 1 * 512);
  _Float16* mw4t = halloc((size_t)1 * 25 * 512);
  _Float16* mw3t = halloc((size_t)13 * 25 * 512);
  _Float16* mw2t = halloc((size_t)13 * 25 * 512);
  _Float16* mw1t = halloc((size_t)13 * 1 * 512);
  _Float16* vw1  = halloc((size_t)1 * 19 * 512);
  _Float16* vw2  = halloc((size_t)10 * 19 * 512);
  _Float16* vw3t = halloc((size_t)1 * 19 * 512);
  _Float16* vw2t = halloc((size_t)10 * 19 * 512);
  _Float16* vw1t = halloc((size_t)10 * 1 * 512);
  _Float16* gw1  = halloc((size_t)1 * 19 * 512);
  _Float16* gw2  = halloc((size_t)10 * 19 * 512);
  _Float16* gw3  = halloc((size_t)10 * 1 * 512);
  float* bm1 = falloc(400); float* bm2 = falloc(400);
  float* bm3 = falloc(400); float* bm4 = falloc(16);
  float* bv1 = falloc(304); float* bv2 = falloc(304);
  float* bg1 = falloc(304); float* bg2 = falloc(304); float* bg3 = falloc(16);
  float* partial = falloc(256);

  auto packW = [&](const float* W, int wr, int wc, int tr, int kt, int nt,
                   _Float16* dst) {
    int total = kt * nt * 512;
    pack_frag<<<(total + 255) / 256, 256, 0, stream>>>(W, wr, wc, tr, kt, nt, dst);
  };
  packW(MW[0],   4, 400, 0,  1, 25, mw1);
  packW(MW[1], 400, 400, 0, 13, 25, mw2);
  packW(MW[2], 400, 400, 0, 13, 25, mw3);
  packW(MW[3], 400,   3, 0, 13,  1, mw4);
  packW(MW[3], 400,   3, 1,  1, 25, mw4t);
  packW(MW[2], 400, 400, 1, 13, 25, mw3t);
  packW(MW[1], 400, 400, 1, 13, 25, mw2t);
  packW(MW[0],   4, 400, 1, 13,  1, mw1t);
  packW(VW[0],   4, 300, 0,  1, 19, vw1);
  packW(VW[1], 300, 300, 0, 10, 19, vw2);
  packW(VW[2], 300,   1, 1,  1, 19, vw3t);
  packW(VW[1], 300, 300, 1, 10, 19, vw2t);
  packW(VW[0],   4, 300, 1, 10,  1, vw1t);
  packW(GW[0],   4, 300, 0,  1, 19, gw1);
  packW(GW[1], 300, 300, 0, 10, 19, gw2);
  packW(GW[2], 300,   4, 0, 10,  1, gw3);

  auto packB = [&](const float* b, int nact, int npad, float* dst) {
    pack_bias<<<(npad + 255) / 256, 256, 0, stream>>>(b, nact, npad, dst);
  };
  packB(Mb[0], 400, 400, bm1); packB(Mb[1], 400, 400, bm2);
  packB(Mb[2], 400, 400, bm3); packB(Mb[3],   3,  16, bm4);
  packB(Vb[0], 300, 304, bv1); packB(Vb[1], 300, 304, bv2);
  packB(Gb[0], 300, 304, bg1); packB(Gb[1], 300, 304, bg2);
  packB(Gb[2],   4,  16, bg3);

  Params prm;
  prm.x = x; prm.times = times;
  prm.mw1 = mw1; prm.mw2 = mw2; prm.mw3 = mw3; prm.mw4 = mw4;
  prm.mw4t = mw4t; prm.mw3t = mw3t; prm.mw2t = mw2t; prm.mw1t = mw1t;
  prm.vw1 = vw1; prm.vw2 = vw2; prm.vw3t = vw3t; prm.vw2t = vw2t; prm.vw1t = vw1t;
  prm.gw1 = gw1; prm.gw2 = gw2; prm.gw3 = gw3;
  prm.bm1 = bm1; prm.bm2 = bm2; prm.bm3 = bm3; prm.bm4 = bm4;
  prm.bv1 = bv1; prm.bv2 = bv2; prm.bg1 = bg1; prm.bg2 = bg2; prm.bg3 = bg3;
  prm.partial = partial;

  symode_main<<<256, 256, SMEM_BYTES, stream>>>(prm);
  reduce_partials<<<1, 32, 0, stream>>>(partial, (float*)d_out, 256,
                                        1.0f / (4096.0f * 10.0f * 8.0f));
}
